// VoxelProjector_32427003085273
// MI455X (gfx1250) — matrix operand order repaired and checked by reading, not compiled, not run on hardware
//
#include <hip/hip_runtime.h>
#include <hip/hip_bf16.h>

typedef __attribute__((ext_vector_type(16))) _Float16 v16h;
typedef __attribute__((ext_vector_type(8)))  _Float16 v8h;
typedef __attribute__((ext_vector_type(8)))  float    v8f;
typedef __attribute__((ext_vector_type(4)))  float    v4f;   // clang vector (nontemporal-store friendly)

// Problem constants (fixed by setup_inputs)
constexpr int B_  = 2, A_ = 4, NPTS = 50000, P_ = 8, L_ = 5;
constexpr int C_  = 256, H_ = 256, W_ = 256;
constexpr int HID = 64, OUTF = 256;
constexpr int W2T_STRIDE   = 72;                 // padded halves per row (144B, 16B-aligned)
constexpr int TILES_PER_BA = NPTS / 16;          // 3125
constexpr int TOTAL_TILES  = B_ * A_ * TILES_PER_BA; // 25000

__device__ __forceinline__ void gatomic_add_f32(float* p, float v) {
    // no-return FP32 global atomic (avoids CAS-loop expansion)
    asm volatile("global_atomic_add_f32 %0, %1, off" :: "v"(p), "v"(v) : "memory");
}

__global__ __launch_bounds__(256) void zero_out_kernel(v4f* __restrict__ p, long long n4) {
    long long i = (long long)blockIdx.x * blockDim.x + threadIdx.x;
    long long stride = (long long)gridDim.x * blockDim.x;
    v4f z = {0.f, 0.f, 0.f, 0.f};
    for (; i < n4; i += stride)
        __builtin_nontemporal_store(z, p + i);
}

__global__ __launch_bounds__(256) void voxel_proj_kernel(
    const float* __restrict__ voxels,   // (B,A,N,P,4)
    const float* __restrict__ coords,   // (B,A,N,4)  [0,x,y,z]
    const float* __restrict__ tmat,     // (B,L,L,4,4)
    const float* __restrict__ W1,       // (4,64)
    const float* __restrict__ b1,       // (64)
    const float* __restrict__ W2,       // (64,256)
    const float* __restrict__ b2,       // (256)
    float* __restrict__ out)            // (B,A,C,H,W)
{
    __shared__ __align__(16) _Float16 sW2T[OUTF * W2T_STRIDE]; // [n][k], transposed, padded
    __shared__ float sW1[4 * HID];
    __shared__ float sB1[HID];
    __shared__ float sB2[OUTF];

    const int tid = threadIdx.x;

    // ---- stage constants in LDS (coalesced global reads) ----
    for (int i = tid; i < 4 * HID; i += 256) sW1[i] = W1[i];
    if (tid < HID) sB1[tid] = b1[tid];
    sB2[tid] = b2[tid];
    for (int i = tid; i < HID * OUTF; i += 256) {
        int k = i >> 8, n = i & 255;                 // W2 row-major (k,n)
        sW2T[n * W2T_STRIDE + k] = (_Float16)W2[i];
    }
    __syncthreads();

    const int wave = blockIdx.x * (blockDim.x >> 5) + (tid >> 5);
    const int lane = tid & 31;
    const int m    = lane & 15;        // point row within tile
    const int hi   = lane >> 4;        // lane-half
    if (wave >= TOTAL_TILES) return;   // wave-uniform

    const int ba = wave / TILES_PER_BA;
    const int n0 = (wave - ba * TILES_PER_BA) * 16;
    const int b  = ba >> 2, a = ba & 3;
    const long long pIdx = (long long)ba * NPTS + n0 + m;

    // ---- vf = mean over P=8 (split across lane halves, combine via shfl_xor 16) ----
    const v4f* vp = (const v4f*)(voxels + (pIdx * P_ + hi * 4) * 4);
    v4f acc = {0.f, 0.f, 0.f, 0.f};
#pragma unroll
    for (int pp = 0; pp < 4; ++pp)
        acc += vp[pp];
    acc.x += __shfl_xor(acc.x, 16, 32);
    acc.y += __shfl_xor(acc.y, 16, 32);
    acc.z += __shfl_xor(acc.z, 16, 32);
    acc.w += __shfl_xor(acc.w, 16, 32);
    const float vfx = acc.x * 0.125f, vfy = acc.y * 0.125f,
                vfz = acc.z * 0.125f, vfw = acc.w * 0.125f;

    // ---- coordinate transform -> output row base ----
    const v4f cxyz = *(const v4f*)(coords + pIdx * 4);         // [pad, x, y, z]
    const float* T = tmat + (b * (L_ * L_) + a) * 16;          // t_matrix[b,0,a]
    const float ex = T[0] * cxyz.y + T[1] * cxyz.z + T[2] * cxyz.w + T[3];
    const float ey = T[4] * cxyz.y + T[5] * cxyz.z + T[6] * cxyz.w + T[7];
    const int xi = min(max((int)(ex * 2.5f), 0), W_ - 1);      // /VOXEL_SIZE, trunc, clip
    const int yi = min(max((int)(ey * 2.5f), 0), H_ - 1);
    const unsigned row = ((unsigned)(ba * C_ + yi)) * H_ + xi;

    // broadcast row bases for the 8 M-rows this lane-half owns in the D-fragment
    unsigned rb[8];
    const int baseLane = hi ? 8 : 0;
#pragma unroll
    for (int r = 0; r < 8; ++r) rb[r] = __shfl(row, baseLane + r, 32);

    // ---- GEMM1 (K=4) computed straight into WMMA A-fragment layout ----
    // lane L holds M=L&15, K in {kOff..kOff+7} U {16+kOff..16+kOff+7}, kOff = hi?8:0
    const int kOff = hi ? 8 : 0;
    v16h afrag0, afrag1;
#pragma unroll
    for (int i = 0; i < 16; ++i) {
        const int kk = (i < 8) ? (kOff + i) : (16 + kOff + (i - 8));
        const int j0 = kk, j1 = 32 + kk;
        float h0 = sB1[j0] + vfx * sW1[j0] + vfy * sW1[HID + j0]
                          + vfz * sW1[2 * HID + j0] + vfw * sW1[3 * HID + j0];
        float h1 = sB1[j1] + vfx * sW1[j1] + vfy * sW1[HID + j1]
                          + vfz * sW1[2 * HID + j1] + vfw * sW1[3 * HID + j1];
        afrag0[i] = (_Float16)fmaxf(h0, 0.f);
        afrag1[i] = (_Float16)fmaxf(h1, 0.f);
    }

    // ---- GEMM2: hidden(16x64) @ W2(64x256) via v_wmma_f32_16x16x32_f16 ----
    const int kOffB = hi ? 8 : 0;      // B-fragment: lane holds N=m, K in {kOffB..kOffB+7} U {16+kOffB..16+kOffB+7}
#pragma unroll 1
    for (int nb = 0; nb < 16; ++nb) {
        const int n = nb * 16 + m;
        const _Float16* wp = &sW2T[n * W2T_STRIDE];

        v16h bfrag0, bfrag1;
        ((v8h*)&bfrag0)[0] = *(const v8h*)(wp + kOffB);
        ((v8h*)&bfrag0)[1] = *(const v8h*)(wp + kOffB + 16);
        ((v8h*)&bfrag1)[0] = *(const v8h*)(wp + 32 + kOffB);
        ((v8h*)&bfrag1)[1] = *(const v8h*)(wp + 32 + kOffB + 16);

        const float bias = sB2[n];
        v8f c;
#pragma unroll
        for (int r = 0; r < 8; ++r) c[r] = bias;   // fuse b2 (added once per point/feature)

        c = __builtin_amdgcn_wmma_f32_16x16x32_f16(false, afrag0, false, bfrag0,
                                                   (short)0, c, false, false);
        c = __builtin_amdgcn_wmma_f32_16x16x32_f16(false, afrag1, false, bfrag1,
                                                   (short)0, c, false, false);

        // D-fragment: lane holds N=n, VGPR r -> M = r + 8*hi
#pragma unroll
        for (int r = 0; r < 8; ++r)
            gatomic_add_f32(out + ((size_t)rb[r] * 256u + (unsigned)n), c[r]);
    }
}

extern "C" void kernel_launch(void* const* d_in, const int* in_sizes, int n_in,
                              void* d_out, int out_size, void* d_ws, size_t ws_size,
                              hipStream_t stream) {
    // dict order: bev_feat, record_len, sparse_voxels, sparse_coords, t_matrix, W1, b1, W2, b2
    const float* voxels = (const float*)d_in[2];
    const float* coords = (const float*)d_in[3];
    const float* tmat   = (const float*)d_in[4];
    const float* W1     = (const float*)d_in[5];
    const float* b1     = (const float*)d_in[6];
    const float* W2     = (const float*)d_in[7];
    const float* b2     = (const float*)d_in[8];
    float* out = (float*)d_out;

    const long long n4 = (long long)out_size / 4;  // out is f32, zero as float4
    zero_out_kernel<<<4096, 256, 0, stream>>>((v4f*)out, n4);
    voxel_proj_kernel<<<TOTAL_TILES / 8, 256, 0, stream>>>(
        voxels, coords, tmat, W1, b1, W2, b2, out);
}
